// WTM_A_29300266894024
// MI455X (gfx1250) — compile-verified, matrix operand order checked by reading, not run on hardware
//
#include <hip/hip_runtime.h>
#include <hip/hip_bf16.h>
#include <math.h>

// ---------------------------------------------------------------------------
// NAT transformer pipeline for gfx1250 (MI455X).
// All large GEMMs use v_wmma_f32_16x16x32_bf16 with bf16 operands staged in a
// bf16 workspace arena (converted ONCE per launch), so the GEMM inner loop is
// pure {2x b128 load + wmma} with zero per-element conversion VALU work.
// Each wave computes a 16x32 output strip (A fragment reused across 2 WMMAs).
// Residual hidden state H stays f32; epilogues fused into the GEMMs.
// Workspace requirement: ~35 MB (see layout in kernel_launch).
// ---------------------------------------------------------------------------

typedef __attribute__((ext_vector_type(16))) __bf16 v16bf;
typedef __attribute__((ext_vector_type(8)))  __bf16 v8bf;
typedef __attribute__((ext_vector_type(8)))  float  v8f;

#define NPIX   1728     // 48*36
#define HDIM   48
#define WDIM   36
#define DIM    256

static __device__ __forceinline__ __bf16 f2bf(float f) {
    unsigned int u = __builtin_bit_cast(unsigned int, f);
    u += 0x7FFFu + ((u >> 16) & 1u);               // RNE
    return __builtin_bit_cast(__bf16, (unsigned short)(u >> 16));
}
static __device__ __forceinline__ float bf2f(__bf16 h) {
    unsigned int x = ((unsigned int)__builtin_bit_cast(unsigned short, h)) << 16;
    return __builtin_bit_cast(float, x);
}
static __device__ __forceinline__ v16bf zero16() {
    v16bf z;
#pragma unroll
    for (int j = 0; j < 16; ++j) z[j] = __builtin_bit_cast(__bf16, (unsigned short)0);
    return z;
}

// Per-lane operand fragment for 16-bit WMMA, K-step of 32 (ISA A 16x32 layout):
// lanes 0-15 hold M=lane, K = k0..k0+7 / k0+16..k0+23; lanes 16-31 hold the
// complementary K-halves. Caller passes p = row + k0 + ((lane&16)?8:0).
// B (32x16 from row-major W[N][K]) uses the mirrored layout with lane&15 = N.
// p is 16-byte aligned (rows are multiples of 32 elements).
static __device__ __forceinline__ v16bf load16bf(const __bf16* __restrict__ p) {
    v8bf lo = *(const v8bf*)(p);
    v8bf hi = *(const v8bf*)(p + 16);
    return __builtin_shufflevector(lo, hi, 0, 1, 2, 3, 4, 5, 6, 7,
                                   8, 9, 10, 11, 12, 13, 14, 15);
}

static __device__ __forceinline__ v8f wmma_bf16(v16bf a, v16bf b, v8f c) {
    return __builtin_amdgcn_wmma_f32_16x16x32_bf16(false, a, false, b, (short)0, c,
                                                   false, false);
}

// ---------------------------------------------------------------------------
// Generic GEMM: out[M x N] = epilogue(A[M x K] * W[N x K]^T)
// grid = (M/16, N/128), block = 128 (4 waves; each wave a 16x32 strip).
// EPI: 0 = +bias -> bf16 out
//      1 = BN+ReLU (p0 scale, p1 shift) -> bf16 out
//      2 = +bias, exact GELU -> bf16 out
//      3 = layer-scale residual: f32 out[m,n] += p0[n]*(acc + bias[n])
// ---------------------------------------------------------------------------
template <int EPI>
__global__ void gemm_wmma(const __bf16* __restrict__ A, int lda,
                          const __bf16* __restrict__ W, int Kdim,
                          const float* __restrict__ bias,
                          const float* __restrict__ p0,
                          const float* __restrict__ p1,
                          void* __restrict__ outv, int ldo) {
    int wave = threadIdx.x >> 5;
    int lane = threadIdx.x & 31;
    int mt   = blockIdx.x;
    int ntb  = (blockIdx.y * 4 + wave) * 2;        // first of two 16-col tiles
    int off  = (lane & 16) ? 8 : 0;
    const __bf16* arow  = A + (size_t)(mt * 16 + (lane & 15)) * lda;
    const __bf16* brow0 = W + (size_t)(ntb * 16 + (lane & 15)) * Kdim;
    const __bf16* brow1 = brow0 + (size_t)16 * Kdim;
    v8f acc0 = {}, acc1 = {};
    for (int k0 = 0; k0 < Kdim; k0 += 32) {
        v16bf a  = load16bf(arow + k0 + off);
        v16bf b0 = load16bf(brow0 + k0 + off);
        v16bf b1 = load16bf(brow1 + k0 + off);
        acc0 = wmma_bf16(a, b0, acc0);
        acc1 = wmma_bf16(a, b1, acc1);
    }
    int mb = mt * 16 + ((lane & 16) ? 8 : 0);
#pragma unroll
    for (int tj = 0; tj < 2; ++tj) {
        v8f acc = tj ? acc1 : acc0;
        int n = (ntb + tj) * 16 + (lane & 15);
        if constexpr (EPI == 0) {
            __bf16* out = (__bf16*)outv;
            float bb = bias[n];
#pragma unroll
            for (int r = 0; r < 8; ++r)
                out[(size_t)(mb + r) * ldo + n] = f2bf(acc[r] + bb);
        } else if constexpr (EPI == 1) {
            __bf16* out = (__bf16*)outv;
            float s = p0[n], sh = p1[n];
#pragma unroll
            for (int r = 0; r < 8; ++r)
                out[(size_t)(mb + r) * ldo + n] = f2bf(fmaxf(acc[r] * s + sh, 0.f));
        } else if constexpr (EPI == 2) {
            __bf16* out = (__bf16*)outv;
            float bb = bias[n];
#pragma unroll
            for (int r = 0; r < 8; ++r) {
                float x = acc[r] + bb;
                out[(size_t)(mb + r) * ldo + n] =
                    f2bf(0.5f * x * (1.f + erff(x * 0.70710678118654752f)));
            }
        } else {
            float* out = (float*)outv;
            float bb = bias[n], g = p0[n];
#pragma unroll
            for (int r = 0; r < 8; ++r) {
                size_t idx = (size_t)(mb + r) * ldo + n;
                out[idx] += g * (acc[r] + bb);
            }
        }
    }
}

// ---------------------------------------------------------------------------
// conv1: X[p][d] = relu(bn1( sum_c concat(in0..in3)[c][p] * W[d][c] ))
// Inputs NCHW f32 (stride-1728 gather, converted here once); W pre-converted
// bf16. K = 384 = 4*96; each 32-wide K block lies in exactly one input.
// ---------------------------------------------------------------------------
__global__ void conv1_wmma(const float* __restrict__ in0, const float* __restrict__ in1,
                           const float* __restrict__ in2, const float* __restrict__ in3,
                           const __bf16* __restrict__ W,
                           const float* __restrict__ scale, const float* __restrict__ shift,
                           float* __restrict__ X) {
    int wave = threadIdx.x >> 5;
    int lane = threadIdx.x & 31;
    int mt   = blockIdx.x;
    int ntb  = (blockIdx.y * 4 + wave) * 2;
    int off  = (lane & 16) ? 8 : 0;
    int m    = mt * 16 + (lane & 15);
    const __bf16* brow0 = W + (size_t)(ntb * 16 + (lane & 15)) * 384;
    const __bf16* brow1 = brow0 + (size_t)16 * 384;
    v8f acc0 = {}, acc1 = {};
#pragma unroll
    for (int k0 = 0; k0 < 384; k0 += 32) {
        const float* inp = (k0 < 96) ? in0 : (k0 < 192) ? in1 : (k0 < 288) ? in2 : in3;
        int cb = k0 - (k0 / 96) * 96;
        v16bf a;
#pragma unroll
        for (int j = 0; j < 8; ++j)
            a[j] = f2bf(inp[(size_t)(cb + off + j) * NPIX + m]);
#pragma unroll
        for (int j = 0; j < 8; ++j)
            a[8 + j] = f2bf(inp[(size_t)(cb + off + 16 + j) * NPIX + m]);
        v16bf b0 = load16bf(brow0 + k0 + off);
        v16bf b1 = load16bf(brow1 + k0 + off);
        acc0 = wmma_bf16(a, b0, acc0);
        acc1 = wmma_bf16(a, b1, acc1);
    }
    int mb = mt * 16 + ((lane & 16) ? 8 : 0);
#pragma unroll
    for (int tj = 0; tj < 2; ++tj) {
        v8f acc = tj ? acc1 : acc0;
        int n = (ntb + tj) * 16 + (lane & 15);
        float s = scale[n], sh = shift[n];
#pragma unroll
        for (int r = 0; r < 8; ++r)
            X[(size_t)(mb + r) * DIM + n] = fmaxf(acc[r] * s + sh, 0.f);
    }
}

// ---------------------------------------------------------------------------
// Direct 3x3 conv via 9-tap WMMA accumulation; bf16 in, weights [tap][O][C].
// FINAL=1 stores f32 NCHW into d_out; else bf16 NHWC.
// ---------------------------------------------------------------------------
template <bool FINAL>
__global__ void conv3x3_wmma(const __bf16* __restrict__ in, const __bf16* __restrict__ wT,
                             const float* __restrict__ scale, const float* __restrict__ shift,
                             void* __restrict__ outv) {
    int wave = threadIdx.x >> 5;
    int lane = threadIdx.x & 31;
    int mt   = blockIdx.x;
    int ntb  = (blockIdx.y * 4 + wave) * 2;
    int off  = (lane & 16) ? 8 : 0;
    int m    = mt * 16 + (lane & 15);
    int y = m / WDIM, x = m % WDIM;
    v8f acc0 = {}, acc1 = {};
#pragma unroll
    for (int t = 0; t < 9; ++t) {
        int yy = y + t / 3 - 1;
        int xx = x + t % 3 - 1;
        bool valid = (yy >= 0) && (yy < HDIM) && (xx >= 0) && (xx < WDIM);
        const __bf16* arow  = in + (size_t)(yy * WDIM + xx) * DIM;
        const __bf16* brow0 = wT + ((size_t)t * DIM + ntb * 16 + (lane & 15)) * DIM;
        const __bf16* brow1 = brow0 + (size_t)16 * DIM;
        for (int k0 = 0; k0 < DIM; k0 += 32) {
            v16bf a = valid ? load16bf(arow + k0 + off) : zero16();
            v16bf b0 = load16bf(brow0 + k0 + off);
            v16bf b1 = load16bf(brow1 + k0 + off);
            acc0 = wmma_bf16(a, b0, acc0);
            acc1 = wmma_bf16(a, b1, acc1);
        }
    }
    int mb = mt * 16 + ((lane & 16) ? 8 : 0);
#pragma unroll
    for (int tj = 0; tj < 2; ++tj) {
        v8f acc = tj ? acc1 : acc0;
        int n = (ntb + tj) * 16 + (lane & 15);
        float s = scale[n], sh = shift[n];
#pragma unroll
        for (int r = 0; r < 8; ++r) {
            float v = fmaxf(acc[r] * s + sh, 0.f);
            int mm = mb + r;
            if constexpr (FINAL)
                ((float*)outv)[(size_t)n * NPIX + mm] = v;
            else
                ((__bf16*)outv)[(size_t)mm * DIM + n] = f2bf(v);
        }
    }
}

// ---------------------------------------------------------------------------
// LayerNorm over C=256; one block per pixel; biased variance; bf16 output.
// ---------------------------------------------------------------------------
__global__ void layernorm_k(const float* __restrict__ in, const float* __restrict__ w,
                            const float* __restrict__ b, __bf16* __restrict__ out) {
    __shared__ float s1[DIM], s2[DIM];
    int p = blockIdx.x, c = threadIdx.x;
    float v = in[(size_t)p * DIM + c];
    s1[c] = v;
    s2[c] = v * v;
    __syncthreads();
    for (int st = 128; st > 0; st >>= 1) {
        if (c < st) { s1[c] += s1[c + st]; s2[c] += s2[c + st]; }
        __syncthreads();
    }
    float mean = s1[0] * (1.f / DIM);
    float var  = s2[0] * (1.f / DIM) - mean * mean;
    float r    = rsqrtf(var + 1e-5f);
    out[(size_t)p * DIM + c] = f2bf((v - mean) * r * w[c] + b[c]);
}

// ---------------------------------------------------------------------------
// NATTEN boundary logic (exact translation of reference _starts).
// ---------------------------------------------------------------------------
static __device__ __forceinline__ void natten_start(int i, int L, int d, int& s, int& p) {
    const int nh = 3, K7 = 7;
    if (d <= 1) {
        s = i - nh; if (s < 0) s = 0;
        if (i + nh >= L) s += L - i - nh - 1;
        p = nh;
        if (i < nh)      p += nh - i;
        if (i + nh >= L) p += L - i - 1 - nh;
    } else if (i - nh * d < 0) {
        s = i % d;
        p = K7 - 1 - i / d;
    } else if (i + nh * d >= L) {
        int imodd = i % d;
        int a = (L / d) * d;
        int b = L - a;
        s = (imodd < b) ? (a + imodd - (K7 - 1) * d) : (a + imodd - K7 * d);
        p = (L - 1 - i) / d;
    } else {
        s = i - nh * d;
        p = nh;
    }
}

// One thread per (pixel, head): flash-style online softmax over the 7x7 window.
__global__ void nat_attn_k(const __bf16* __restrict__ qkv, const float* __restrict__ rpb,
                           __bf16* __restrict__ outp, int dil) {
    int t = blockIdx.x * blockDim.x + threadIdx.x;
    if (t >= NPIX * 8) return;
    int head = t & 7;
    int pix  = t >> 3;
    int y = pix / WDIM, x = pix % WDIM;
    int si, pi, sj, pj;
    natten_start(y, HDIM, dil, si, pi);
    natten_start(x, WDIM, dil, sj, pj);
    const int hc = head * 32;
    const __bf16* qp = qkv + (size_t)pix * 768 + hc;
    float q[32];
#pragma unroll
    for (int d = 0; d < 32; ++d) q[d] = bf2f(qp[d]) * 0.17677669529663687f;  // 32^-0.5
    float m = -1e30f, ssum = 0.f;
    float o[32];
#pragma unroll
    for (int d = 0; d < 32; ++d) o[d] = 0.f;
    for (int ki = 0; ki < 7; ++ki) {
        int ny = si + ki * dil;
        const float* biasrow = rpb + ((size_t)head * 13 + (pi + ki)) * 13 + pj;
        for (int kj = 0; kj < 7; ++kj) {
            int nx = sj + kj * dil;
            const __bf16* kvp = qkv + (size_t)(ny * WDIM + nx) * 768 + 256 + hc;
            float sc = biasrow[kj];
#pragma unroll
            for (int d = 0; d < 32; ++d) sc += q[d] * bf2f(kvp[d]);
            float nm = fmaxf(m, sc);
            float cf = expf(m - nm);
            float pf = expf(sc - nm);
            ssum = ssum * cf + pf;
            const __bf16* vp = kvp + 256;
#pragma unroll
            for (int d = 0; d < 32; ++d) o[d] = o[d] * cf + pf * bf2f(vp[d]);
            m = nm;
        }
    }
    float inv = 1.f / ssum;
    __bf16* op = outp + (size_t)pix * DIM + hc;
#pragma unroll
    for (int d = 0; d < 32; ++d) op[d] = f2bf(o[d] * inv);
}

// ---------------------------------------------------------------------------
// Small utility kernels.
// ---------------------------------------------------------------------------
__global__ void bn_prep(const float* __restrict__ p, float* __restrict__ scale,
                        float* __restrict__ shift) {
    int c = threadIdx.x;
    float s = p[c], b = p[256 + c], m = p[512 + c], v = p[768 + c];
    float sc = s * rsqrtf(v + 1e-5f);
    scale[c] = sc;
    shift[c] = b - m * sc;
}

__global__ void cvt_bf16(const float* __restrict__ s, __bf16* __restrict__ d, int n) {
    int i = blockIdx.x * blockDim.x + threadIdx.x;
    if (i < n) d[i] = f2bf(s[i]);
}

__global__ void repack3x3(const float* __restrict__ w, __bf16* __restrict__ wT) {
    int idx = blockIdx.x * blockDim.x + threadIdx.x;
    if (idx >= 256 * 256 * 9) return;
    int t = idx % 9;
    int c = (idx / 9) % 256;
    int o = idx / (9 * 256);
    wT[((size_t)t * 256 + o) * 256 + c] = f2bf(w[idx]);  // w is OIHW [o][c][ky][kx]
}

__global__ void col_mean(const float* __restrict__ X, float* __restrict__ gm) {
    int d = threadIdx.x;
    float s = 0.f;
    for (int p = 0; p < NPIX; ++p) s += X[(size_t)p * DIM + d];
    gm[d] = s * (1.f / NPIX);
}

__global__ void gap_vec(const float* __restrict__ gm, const float* __restrict__ gw,
                        const float* __restrict__ scale, const float* __restrict__ shift,
                        float* __restrict__ gv) {
    int d = threadIdx.x;
    float acc = 0.f;
    for (int c = 0; c < DIM; ++c) acc += gm[c] * gw[(size_t)d * DIM + c];
    gv[d] = fmaxf(acc * scale[d] + shift[d], 0.f);
}

__global__ void copy_cat(const float* __restrict__ h, __bf16* __restrict__ cat, int colOff) {
    int e = blockIdx.x * blockDim.x + threadIdx.x;
    int p = e >> 8, c = e & 255;
    cat[(size_t)p * 1280 + colOff + c] = f2bf(h[e]);
}

__global__ void fill_x5(const float* __restrict__ gv, __bf16* __restrict__ cat) {
    int e = blockIdx.x * blockDim.x + threadIdx.x;
    int p = e >> 8, c = e & 255;
    cat[(size_t)p * 1280 + 1024 + c] = f2bf(gv[c]);
}

// ---------------------------------------------------------------------------
extern "C" void kernel_launch(void* const* d_in, const int* in_sizes, int n_in,
                              void* d_out, int out_size, void* d_ws, size_t ws_size,
                              hipStream_t stream) {
    (void)in_sizes; (void)n_in; (void)out_size; (void)ws_size;
    const float* in0     = (const float*)d_in[0];
    const float* in1     = (const float*)d_in[1];
    const float* in2     = (const float*)d_in[2];
    const float* in3     = (const float*)d_in[3];
    const float* conv1_w = (const float*)d_in[4];
    const float* bn1     = (const float*)d_in[5];
    const float* ln1_w   = (const float*)d_in[6];
    const float* ln1_b   = (const float*)d_in[7];
    const float* qkv_w   = (const float*)d_in[8];
    const float* qkv_b   = (const float*)d_in[9];
    const float* rpb     = (const float*)d_in[10];
    const float* proj_w  = (const float*)d_in[11];
    const float* proj_b  = (const float*)d_in[12];
    const float* gamma1  = (const float*)d_in[13];
    const float* ln2_w   = (const float*)d_in[14];
    const float* ln2_b   = (const float*)d_in[15];
    const float* fc1_w   = (const float*)d_in[16];
    const float* fc1_b   = (const float*)d_in[17];
    const float* fc2_w   = (const float*)d_in[18];
    const float* fc2_b   = (const float*)d_in[19];
    const float* gamma2  = (const float*)d_in[20];
    const float* gap_w   = (const float*)d_in[21];
    const float* gap_bn  = (const float*)d_in[22];
    const float* conv2_w = (const float*)d_in[23];
    const float* bn2     = (const float*)d_in[24];
    const float* lc1_w   = (const float*)d_in[25];
    const float* lbn1    = (const float*)d_in[26];
    const float* lc2_w   = (const float*)d_in[27];
    const float* lbn2    = (const float*)d_in[28];

    // ---- Workspace layout: f32 region then bf16 arena (~35 MB total) ----
    float* ws  = (float*)d_ws;
    float* X   = ws;                 // 1728*256  conv1 output (kept for GAP)
    float* H   = X + 442368;         // 1728*256  running hidden state (f32)
    float* BNS = H + 442368;         // 5*(256 scale + 256 shift)
    float* GM  = BNS + 2560;         // 256
    float* GV  = GM + 256;           // 256
    __bf16* bb   = (__bf16*)(GV + 256);
    __bf16* LNB  = bb;  bb += 442368;    // layernorm output
    __bf16* QKVb = bb;  bb += 1327104;   // 1728*768
    __bf16* ATTb = bb;  bb += 442368;
    __bf16* MIDb = bb;  bb += 1769472;   // 1728*1024
    __bf16* CATb = bb;  bb += 2211840;   // 1728*1280
    __bf16* C2Ob = bb;  bb += 442368;
    __bf16* L1Ob = bb;  bb += 442368;
    __bf16* WT1b = bb;  bb += 589824;    // repacked lc1_w [9][256][256]
    __bf16* WT2b = bb;  bb += 589824;
    __bf16* W1b  = bb;  bb += 98304;     // conv1_w bf16
    __bf16* QWb  = bb;  bb += 1572864;   // qkv_w bf16 (8 layers)
    __bf16* PWb  = bb;  bb += 524288;    // proj_w bf16
    __bf16* F1Wb = bb;  bb += 2097152;   // fc1_w bf16
    __bf16* F2Wb = bb;  bb += 2097152;   // fc2_w bf16
    __bf16* C2Wb = bb;  bb += 327680;    // conv2_w bf16

    static const int DILS[8] = {1, 2, 3, 4, 5, 4, 2, 1};

    // One-time parameter prep: BN affine folds + weight conversions to bf16.
    bn_prep<<<1, 256, 0, stream>>>(bn1,    BNS + 0,    BNS + 256);
    bn_prep<<<1, 256, 0, stream>>>(bn2,    BNS + 512,  BNS + 768);
    bn_prep<<<1, 256, 0, stream>>>(gap_bn, BNS + 1024, BNS + 1280);
    bn_prep<<<1, 256, 0, stream>>>(lbn1,   BNS + 1536, BNS + 1792);
    bn_prep<<<1, 256, 0, stream>>>(lbn2,   BNS + 2048, BNS + 2304);
    cvt_bf16<<<(98304 + 255) / 256, 256, 0, stream>>>(conv1_w, W1b, 98304);
    cvt_bf16<<<(1572864 + 255) / 256, 256, 0, stream>>>(qkv_w, QWb, 1572864);
    cvt_bf16<<<(524288 + 255) / 256, 256, 0, stream>>>(proj_w, PWb, 524288);
    cvt_bf16<<<(2097152 + 255) / 256, 256, 0, stream>>>(fc1_w, F1Wb, 2097152);
    cvt_bf16<<<(2097152 + 255) / 256, 256, 0, stream>>>(fc2_w, F2Wb, 2097152);
    cvt_bf16<<<(327680 + 255) / 256, 256, 0, stream>>>(conv2_w, C2Wb, 327680);
    repack3x3<<<(589824 + 255) / 256, 256, 0, stream>>>(lc1_w, WT1b);
    repack3x3<<<(589824 + 255) / 256, 256, 0, stream>>>(lc2_w, WT2b);

    // Stem: conv1 + BN + ReLU -> X (f32 NHWC), then H = X.
    conv1_wmma<<<dim3(108, 2), 128, 0, stream>>>(in0, in1, in2, in3, W1b,
                                                 BNS + 0, BNS + 256, X);
    hipMemcpyAsync(H, X, (size_t)442368 * sizeof(float),
                   hipMemcpyDeviceToDevice, stream);

    // GAP branch (reads pre-layer X): x5 = relu(bn(gap_w @ mean(X))).
    col_mean<<<1, 256, 0, stream>>>(X, GM);
    gap_vec<<<1, 256, 0, stream>>>(GM, gap_w, BNS + 1024, BNS + 1280, GV);
    fill_x5<<<NPIX, 256, 0, stream>>>(GV, CATb);

    // 8 transformer layers.
    for (int i = 0; i < 8; ++i) {
        layernorm_k<<<NPIX, 256, 0, stream>>>(H, ln1_w + i * 256, ln1_b + i * 256, LNB);
        gemm_wmma<0><<<dim3(108, 6), 128, 0, stream>>>(
            LNB, 256, QWb + (size_t)i * 196608, 256, qkv_b + i * 768,
            nullptr, nullptr, QKVb, 768);
        nat_attn_k<<<(NPIX * 8) / 256, 256, 0, stream>>>(
            QKVb, rpb + (size_t)i * 1352, ATTb, DILS[i]);
        gemm_wmma<3><<<dim3(108, 2), 128, 0, stream>>>(
            ATTb, 256, PWb + (size_t)i * 65536, 256, proj_b + i * 256,
            gamma1 + i * 256, nullptr, H, 256);
        layernorm_k<<<NPIX, 256, 0, stream>>>(H, ln2_w + i * 256, ln2_b + i * 256, LNB);
        gemm_wmma<2><<<dim3(108, 8), 128, 0, stream>>>(
            LNB, 256, F1Wb + (size_t)i * 262144, 256, fc1_b + i * 1024,
            nullptr, nullptr, MIDb, 1024);
        gemm_wmma<3><<<dim3(108, 2), 128, 0, stream>>>(
            MIDb, 1024, F2Wb + (size_t)i * 262144, 1024, fc2_b + i * 256,
            gamma2 + i * 256, nullptr, H, 256);
        if (i & 1)
            copy_cat<<<NPIX, 256, 0, stream>>>(H, CATb, (i >> 1) * 256);
    }

    // Head: conv2 (1280->256) + BN + ReLU, then two 3x3 convs + BN + ReLU.
    gemm_wmma<1><<<dim3(108, 2), 128, 0, stream>>>(
        CATb, 1280, C2Wb, 1280, nullptr, BNS + 512, BNS + 768, C2Ob, 256);
    conv3x3_wmma<false><<<dim3(108, 2), 128, 0, stream>>>(C2Ob, WT1b, BNS + 1536,
                                                          BNS + 1792, L1Ob);
    conv3x3_wmma<true><<<dim3(108, 2), 128, 0, stream>>>(L1Ob, WT2b, BNS + 2048,
                                                         BNS + 2304, (float*)d_out);
}
